// STCNNT_Cell_35450660061975
// MI455X (gfx1250) — compile-verified, matrix operand order checked by reading, not run on hardware
//
#include <hip/hip_runtime.h>

// MI455X / gfx1250: wave32, WMMA 16x16x32 bf16 (f32 accum).
typedef __attribute__((ext_vector_type(16))) __bf16 v16bf;
typedef __attribute__((ext_vector_type(8)))  float  v8f;

__device__ __forceinline__ v8f wmma_bf16(v16bf a, v16bf b, v8f c) {
  // D = A(16x32) x B(32x16) + C(16x16 f32)
  return __builtin_amdgcn_wmma_f32_16x16x32_bf16(false, a, false, b, (short)0, c,
                                                 false, false);
}

// ---- gfx1250 async global->LDS copy (ASYNCcnt path), with safe fallback ----
#if defined(__HIP_DEVICE_COMPILE__) && \
    __has_builtin(__builtin_amdgcn_global_load_async_to_lds_b128)
#define HAVE_ASYNC_LDS 1
#endif

__device__ __forceinline__ void async_copy_b128(void* lds_ptr, const void* gptr) {
#ifdef HAVE_ASYNC_LDS
  typedef int v4i_pay __attribute__((vector_size(16)));
  typedef __attribute__((address_space(1))) v4i_pay* gp_t;
  typedef __attribute__((address_space(3))) v4i_pay* lp_t;
  gp_t g = (gp_t)(uintptr_t)gptr;
  // flat LDS pointer: low 32 bits are the LDS byte address (ISA 10.2)
  lp_t l = (lp_t)(unsigned int)(uintptr_t)lds_ptr;
  __builtin_amdgcn_global_load_async_to_lds_b128(g, l, 0, 0);
#else
  *(uint4*)lds_ptr = *(const uint4*)gptr;
#endif
}

__device__ __forceinline__ void wait_async_zero() {
#ifdef HAVE_ASYNC_LDS
#if __has_builtin(__builtin_amdgcn_s_wait_asynccnt)
  __builtin_amdgcn_s_wait_asynccnt(0);
#else
  asm volatile("s_wait_asynccnt 0x0" ::: "memory");
#endif
#endif
}

// ---------------------------------------------------------------------------
// Kernel 0: repack conv weights fp32 [O][I][3][3] -> bf16 [t][s][oc][ic]
// (ic contiguous => 32B-aligned WMMA B fragments loaded straight from global)
// ---------------------------------------------------------------------------
__global__ __launch_bounds__(256) void prep_weights(
    const float* __restrict__ wq, const float* __restrict__ wk,
    const float* __restrict__ wv, const float* __restrict__ wo,
    __bf16* __restrict__ qkvw, __bf16* __restrict__ wow) {
  int idx = blockIdx.x * 256 + threadIdx.x;
  const int nqkv = 3 * 9 * 64 * 64;
  if (idx < nqkv) {
    int t = idx / (9 * 4096);
    int r = idx % (9 * 4096);
    int s = r / 4096;
    int r2 = r % 4096;
    int oc = r2 >> 6, ic = r2 & 63;
    const float* src = (t == 0) ? wq : (t == 1) ? wk : wv;
    qkvw[idx] = (__bf16)src[(oc * 64 + ic) * 9 + s];
  } else {
    int j = idx - nqkv;
    if (j < 9 * 4096) {
      int s = j / 4096;
      int r2 = j % 4096;
      int oc = r2 >> 6, ic = r2 & 63;
      wow[j] = (__bf16)wo[(oc * 64 + ic) * 9 + s];
    }
  }
}

// ---------------------------------------------------------------------------
// Kernel 1: im2grid + fused QKV 3x3 conv on zero-padded 8x8 dilated patches.
// One WG = 2 patches. Conv as 9 shifted GEMMs, M=pixels N=192(q|k|v) K=64ic.
// B fragments batch-loaded (12 per k-step) so loads clause up ahead of the
// WMMA burst instead of load->wait(0)->wmma serialization.
// Outputs: Q,K as [bh][token][d], V transposed as [bh][d][token] (bf16).
// ---------------------------------------------------------------------------
__global__ __launch_bounds__(256) void qkv_conv(
    const float* __restrict__ x,
    const float* __restrict__ bq, const float* __restrict__ bk,
    const float* __restrict__ bv,
    const __bf16* __restrict__ qkvw,
    __bf16* __restrict__ qws, __bf16* __restrict__ kws,
    __bf16* __restrict__ vws) {
  extern __shared__ char smem[];
  __bf16* xt = (__bf16*)smem;  // 2 patches x 10x10 halo x 64ch (25600 B)
  const int tid = threadIdx.x;
  const int p0 = blockIdx.x * 2;

  // zero halo buffer
  {
    uint4 z = {0, 0, 0, 0};
    uint4* lz = (uint4*)smem;
    for (int j = tid; j < (2 * 6400 * 2) / 16; j += 256) lz[j] = z;
  }
  __syncthreads();
  // gather dilated patch interiors (h = gi*32+hg, w = gj*32+wg)
  for (int e = tid; e < 2 * 4096; e += 256) {
    int pp = e >> 12;
    int r = e & 4095;
    int c = r >> 6;
    int pix = r & 63;
    int patch = p0 + pp;
    int bt = patch >> 10;
    int hwg = patch & 1023;
    int hg = hwg >> 5, wg = hwg & 31;
    int gi = pix >> 3, gj = pix & 7;
    int h = gi * 32 + hg, w = gj * 32 + wg;
    float v = x[(((size_t)bt * 64 + c) * 256 + h) * 256 + w];
    xt[pp * 6400 + ((gi + 1) * 10 + (gj + 1)) * 64 + c] = (__bf16)v;
  }
  __syncthreads();

  const int wave = tid >> 5, lane = tid & 31;
  const int pp = wave >> 2;            // patch within WG
  const int mloc = (wave & 3) * 16;    // 16-pixel strip
  const int lrow = lane & 15;
  const int khalf = (lane >> 4) * 16;
  const int patch = p0 + pp;
  const int bt = patch >> 10;
  const int token = patch & 1023;
  const int rA = mloc + lrow;
  const int gy = rA >> 3, gx = rA & 7;

  v8f acc[12];
#pragma unroll
  for (int nt = 0; nt < 12; ++nt) {
    int ch = nt * 16 + lrow;
    int tsel = ch >> 6, oc = ch & 63;
    float bias = (tsel == 0) ? bq[oc] : (tsel == 1) ? bk[oc] : bv[oc];
    v8f t = {bias, bias, bias, bias, bias, bias, bias, bias};
    acc[nt] = t;
  }

  const __bf16* xbase = xt + pp * 6400;
  for (int s = 0; s < 9; ++s) {
    int dy = s / 3, dx = s % 3;
    const __bf16* arow = xbase + ((gy + dy) * 10 + (gx + dx)) * 64;
#pragma unroll
    for (int kk = 0; kk < 2; ++kk) {
      int kidx = khalf + kk * 32;
      v16bf a = *(const v16bf*)(arow + kidx);
      // batch all 12 B fragments first (independent regs -> load clause)
      v16bf barr[12];
#pragma unroll
      for (int nt = 0; nt < 12; ++nt) {
        int tsel = nt >> 2;
        int oc = (nt & 3) * 16 + lrow;
        barr[nt] =
            *(const v16bf*)(qkvw + (((size_t)tsel * 9 + s) * 64 + oc) * 64 + kidx);
      }
#pragma unroll
      for (int nt = 0; nt < 12; ++nt) acc[nt] = wmma_bf16(a, barr[nt], acc[nt]);
    }
  }

#pragma unroll
  for (int nt = 0; nt < 12; ++nt) {
    int ch = nt * 16 + lrow;
    int tsel = ch >> 6, oc = ch & 63;
    int hd = oc >> 3, hc = oc & 7;
#pragma unroll
    for (int i = 0; i < 8; ++i) {
      int pix = mloc + i + 8 * (lane >> 4);
      int d = hc * 64 + pix;
      __bf16 vv = (__bf16)acc[nt][i];
      if (tsel == 0)
        qws[(((size_t)(bt * 8 + hd)) * 1024 + token) * 512 + d] = vv;
      else if (tsel == 1)
        kws[(((size_t)(bt * 8 + hd)) * 1024 + token) * 512 + d] = vv;
      else
        vws[(((size_t)(bt * 8 + hd)) * 512 + d) * 1024 + token] = vv;
    }
  }
}

// ---------------------------------------------------------------------------
// Kernel 2: attention per (bt, head, 32-query tile). Full 32x1024 score panel
// resident in LDS (exact softmax), PV accumulated in registers.
// K / V^T blocks streamed global->LDS with ASYNC copies (ASYNCcnt) when the
// toolchain exposes them.
// LDS: Q 32K | K/V^T block 64K (shared) | S 128K | P 64K | rsum  (~288 KB)
// ---------------------------------------------------------------------------
__global__ __launch_bounds__(256) void attn(
    const __bf16* __restrict__ qws, const __bf16* __restrict__ kws,
    const __bf16* __restrict__ vws, __bf16* __restrict__ ybf) {
  extern __shared__ char smem[];
  __bf16* qs = (__bf16*)smem;                                   // 32 x 512
  __bf16* kv = (__bf16*)(smem + 32768);                         // 64x512 / 512x64
  float* S = (float*)(smem + 32768 + 65536);                    // 32 x 1024
  __bf16* P = (__bf16*)(smem + 32768 + 65536 + 131072);         // 32 x 1024
  float* rsum = (float*)(smem + 32768 + 65536 + 131072 + 65536);

  const int tid = threadIdx.x;
  const int wave = tid >> 5, lane = tid & 31;
  const int lrow = lane & 15, khalf = (lane >> 4) * 16;
  const int m0 = blockIdx.x * 32;
  const int head = blockIdx.y, bt = blockIdx.z;
  const size_t bh = (size_t)(bt * 8 + head);
  const float scale = 0.044194173824159216f;  // 1/sqrt(512)

  {  // load Q tile (async global->LDS)
    const uint4* src = (const uint4*)(qws + bh * (1024 * 512) + (size_t)m0 * 512);
    uint4* dst = (uint4*)qs;
    for (int j = tid; j < 2048; j += 256) async_copy_b128(&dst[j], &src[j]);
  }

  // ---- phase 1: S = scale * Q K^T  (each wave: one 16x16 tile per block) ---
  const int mt = wave >> 2, nt = wave & 3;
  for (int kb = 0; kb < 16; ++kb) {
    __syncthreads();
    const uint4* src = (const uint4*)(kws + bh * (1024 * 512) + (size_t)kb * 64 * 512);
    uint4* dst = (uint4*)kv;
    for (int j = tid; j < 4096; j += 256) async_copy_b128(&dst[j], &src[j]);
    wait_async_zero();
    __syncthreads();
    v8f acc = {};
    const __bf16* qrow = qs + (mt * 16 + lrow) * 512 + khalf;
    const __bf16* krow = kv + (nt * 16 + lrow) * 512 + khalf;
#pragma unroll
    for (int ks = 0; ks < 16; ++ks) {
      v16bf a = *(const v16bf*)(qrow + ks * 32);
      v16bf b = *(const v16bf*)(krow + ks * 32);
      acc = wmma_bf16(a, b, acc);
    }
#pragma unroll
    for (int i = 0; i < 8; ++i) {
      int r = mt * 16 + i + 8 * (lane >> 4);
      int cg = kb * 64 + nt * 16 + lrow;
      S[r * 1024 + cg] = acc[i] * scale;
    }
  }
  __syncthreads();

  // ---- softmax: 4 rows per wave, lane-strided + shfl_xor reductions --------
  for (int rr = 0; rr < 4; ++rr) {
    int r = wave * 4 + rr;
    float mx = -3.0e38f;
    for (int j = lane; j < 1024; j += 32) mx = fmaxf(mx, S[r * 1024 + j]);
#pragma unroll
    for (int off = 16; off > 0; off >>= 1) mx = fmaxf(mx, __shfl_xor(mx, off, 32));
    float sum = 0.f;
    for (int j = lane; j < 1024; j += 32) {
      float e = __expf(S[r * 1024 + j] - mx);
      P[r * 1024 + j] = (__bf16)e;
      sum += e;
    }
#pragma unroll
    for (int off = 16; off > 0; off >>= 1) sum += __shfl_xor(sum, off, 32);
    if (lane == 0) rsum[r] = sum;
  }

  // ---- phase 2: O = P V  (wave owns 64 d-cols; V^T blocks streamed) --------
  v8f o[2][4];
#pragma unroll
  for (int a0 = 0; a0 < 2; ++a0)
#pragma unroll
    for (int b0 = 0; b0 < 4; ++b0) {
      v8f z = {};
      o[a0][b0] = z;
    }
  for (int vb = 0; vb < 16; ++vb) {
    __syncthreads();
    uint4* dst = (uint4*)kv;  // layout [d(512)][tok(64)]
    for (int j = tid; j < 4096; j += 256) {
      int d = j >> 3, ci = j & 7;
      const uint4* src =
          (const uint4*)(vws + bh * (512 * 1024) + (size_t)d * 1024 + vb * 64);
      async_copy_b128(&dst[j], &src[ci]);
    }
    wait_async_zero();
    __syncthreads();
#pragma unroll
    for (int mti = 0; mti < 2; ++mti) {
#pragma unroll
      for (int ks = 0; ks < 2; ++ks) {
        v16bf a = *(const v16bf*)(P + (mti * 16 + lrow) * 1024 + vb * 64 + khalf +
                                  ks * 32);
#pragma unroll
        for (int nti = 0; nti < 4; ++nti) {
          int drow = wave * 64 + nti * 16 + lrow;
          v16bf b = *(const v16bf*)(kv + drow * 64 + khalf + ks * 32);
          o[mti][nti] = wmma_bf16(a, b, o[mti][nti]);
        }
      }
    }
  }
  __syncthreads();

  // ---- normalize + grid2im scatter (bf16 channels-last [bt][h][w][c]) ------
#pragma unroll
  for (int mti = 0; mti < 2; ++mti)
#pragma unroll
    for (int nti = 0; nti < 4; ++nti)
#pragma unroll
      for (int i = 0; i < 8; ++i) {
        int r = mti * 16 + i + 8 * (lane >> 4);
        int token = m0 + r;
        int d = wave * 64 + nti * 16 + lrow;
        float val = o[mti][nti][i] / rsum[r];
        int hc = d >> 6, pix = d & 63;
        int gi = pix >> 3, gj = pix & 7;
        int hg = token >> 5, wg = token & 31;
        int c = head * 8 + hc;
        int hh = gi * 32 + hg, ww = gj * 32 + wg;
        ybf[(((size_t)bt * 256 + hh) * 256 + ww) * 64 + c] = (__bf16)val;
      }
}

// ---------------------------------------------------------------------------
// Kernel 3: out = y + conv3x3(y, wo) + bo on full 256x256 image.
// One WG = 16x16 output tile; 18x18x64 halo in LDS (async for interior);
// 9 shifted GEMMs, B fragments hoisted across the two m-tiles.
// ---------------------------------------------------------------------------
__global__ __launch_bounds__(256) void out_conv(
    const __bf16* __restrict__ ybf, const __bf16* __restrict__ wow,
    const float* __restrict__ bo, float* __restrict__ out) {
  extern __shared__ char smem[];
  __bf16* tile = (__bf16*)smem;  // 18*18*64 bf16 (41472 B)
  const int tid = threadIdx.x;
  const int tx = blockIdx.x, ty = blockIdx.y, bt = blockIdx.z;

  uint4* dst = (uint4*)tile;
  for (int q = tid; q < 18 * 18 * 8; q += 256) {
    int pos = q >> 3, ci = q & 7;
    int iy = pos / 18, ix = pos % 18;
    int y = ty * 16 - 1 + iy, x = tx * 16 - 1 + ix;
    if (y >= 0 && y < 256 && x >= 0 && x < 256) {
      const uint4* src = (const uint4*)(ybf + (((size_t)bt * 256 + y) * 256 + x) * 64);
      async_copy_b128(&dst[q], &src[ci]);
    } else {
      uint4 z = {0, 0, 0, 0};
      dst[q] = z;
    }
  }
  wait_async_zero();
  __syncthreads();

  const int wave = tid >> 5, lane = tid & 31;
  const int lrow = lane & 15, khalf = (lane >> 4) * 16;
  v8f acc[2][4];
#pragma unroll
  for (int mti = 0; mti < 2; ++mti)
#pragma unroll
    for (int nti = 0; nti < 4; ++nti) {
      float b = bo[nti * 16 + lrow];
      v8f t = {b, b, b, b, b, b, b, b};
      acc[mti][nti] = t;
    }

  for (int s = 0; s < 9; ++s) {
    int dy = s / 3, dx = s % 3;
#pragma unroll
    for (int kk = 0; kk < 2; ++kk) {
      int kidx = khalf + kk * 32;
      // hoist the 4 weight fragments (shared by both m-tiles)
      v16bf b4[4];
#pragma unroll
      for (int nti = 0; nti < 4; ++nti)
        b4[nti] =
            *(const v16bf*)(wow + ((size_t)s * 64 + nti * 16 + lrow) * 64 + kidx);
#pragma unroll
      for (int mti = 0; mti < 2; ++mti) {
        int m = (wave + mti * 8) * 16 + lrow;
        int my = m >> 4, mx = m & 15;
        v16bf a = *(const v16bf*)(tile + ((my + dy) * 18 + (mx + dx)) * 64 + kidx);
#pragma unroll
        for (int nti = 0; nti < 4; ++nti)
          acc[mti][nti] = wmma_bf16(a, b4[nti], acc[mti][nti]);
      }
    }
  }

#pragma unroll
  for (int mti = 0; mti < 2; ++mti)
#pragma unroll
    for (int nti = 0; nti < 4; ++nti)
#pragma unroll
      for (int i = 0; i < 8; ++i) {
        int m = (wave + mti * 8) * 16 + i + 8 * (lane >> 4);
        int my = m >> 4, mx = m & 15;
        int y = ty * 16 + my, x = tx * 16 + mx;
        int c = nti * 16 + lrow;
        float resid = (float)ybf[(((size_t)bt * 256 + y) * 256 + x) * 64 + c];
        out[(((size_t)bt * 64 + c) * 256 + y) * 256 + x] =
            resid + acc[mti][nti][i];
      }
}

// ---------------------------------------------------------------------------
extern "C" void kernel_launch(void* const* d_in, const int* in_sizes, int n_in,
                              void* d_out, int out_size, void* d_ws,
                              size_t ws_size, hipStream_t stream) {
  const float* x = (const float*)d_in[0];
  const float* wq = (const float*)d_in[1];
  const float* bq = (const float*)d_in[2];
  const float* wk = (const float*)d_in[3];
  const float* bk = (const float*)d_in[4];
  const float* wv = (const float*)d_in[5];
  const float* bv = (const float*)d_in[6];
  const float* wo = (const float*)d_in[7];
  const float* bo = (const float*)d_in[8];
  float* out = (float*)d_out;
  char* ws = (char*)d_ws;

  const size_t SZ = (size_t)64 * 1024 * 512 * 2;  // 67,108,864 B per tensor
  __bf16* qws = (__bf16*)(ws);
  __bf16* kws = (__bf16*)(ws + SZ);
  __bf16* vws = (__bf16*)(ws + 2 * SZ);        // V^T: [bh][d][token]
  __bf16* ybf = (__bf16*)(ws + 3 * SZ);        // [bt][h][w][c] bf16
  __bf16* qkvw = (__bf16*)(ws + 4 * SZ);       // 3*9*64*64 bf16
  __bf16* wow = (__bf16*)(ws + 4 * SZ + (size_t)3 * 9 * 64 * 64 * 2);

  prep_weights<<<576, 256, 0, stream>>>(wq, wk, wv, wo, qkvw, wow);
  qkv_conv<<<4096, 256, 25600, stream>>>(x, bq, bk, bv, qkvw, qws, kws, vws);
  attn<<<dim3(32, 8, 8), 256, 295168, stream>>>(qws, kws, vws, ybf);
  out_conv<<<dim3(16, 16, 8), 256, 41472, stream>>>(ybf, wow, bo, out);
}
